// MultiHeadAttention_78838419685826
// MI455X (gfx1250) — matrix operand order repaired and checked by reading, not compiled, not run on hardware
//
#include <hip/hip_runtime.h>
#include <hip/hip_bf16.h>

// Problem constants (from reference)
#define Bb 4
#define Ss 2048
#define Ee 1024
#define Hh 16
#define Dd 64
#define Mm (Bb * Ss)       // 8192 rows of activations
#define HED (Hh * Ee * Dd) // 1048576
#define BHSD ((size_t)Bb * Hh * Ss * Dd)
#define BSE ((size_t)Bb * Ss * Ee)

typedef __bf16 bf16;
typedef __attribute__((ext_vector_type(16))) __bf16 v16bf;
typedef __attribute__((ext_vector_type(8)))  __bf16 v8bf;
typedef __attribute__((ext_vector_type(8)))  float  v8f;

// ---- optional CDNA5 async global->LDS path (guarded; falls back to sync) ----
#if defined(__has_builtin)
#if __has_builtin(__builtin_amdgcn_global_load_async_to_lds_b128) && \
    __has_builtin(__builtin_amdgcn_s_wait_asynccnt)
#define HAVE_ASYNC_LDS 1
#endif
#endif
#ifndef HAVE_ASYNC_LDS
#define HAVE_ASYNC_LDS 0
#endif

#if HAVE_ASYNC_LDS
#define ASG __attribute__((address_space(1)))
#define ASL __attribute__((address_space(3)))
typedef int gv4i __attribute__((vector_size(16)));  // GCC-style int4 (b128 payload)
__device__ __forceinline__ void async_cp16(const bf16* gsrc, bf16* lds) {
  __builtin_amdgcn_global_load_async_to_lds_b128(
      (ASG gv4i*)gsrc, (ASL gv4i*)lds, /*offset=*/0, /*cpol=*/0);
}
__device__ __forceinline__ void async_wait() {
  __builtin_amdgcn_s_wait_asynccnt(0);
}
#else
__device__ __forceinline__ void async_cp16(const bf16* gsrc, bf16* lds) {
  *(uint4*)lds = *(const uint4*)gsrc;
}
__device__ __forceinline__ void async_wait() {}
#endif

__device__ __forceinline__ bf16 f2bf(float f) { return (bf16)f; }

__device__ __forceinline__ v8f vzero() {
  v8f z;
#pragma unroll
  for (int i = 0; i < 8; ++i) z[i] = 0.0f;
  return z;
}

// Build a 16-element bf16 operand from two contiguous 8-element LDS chunks.
__device__ __forceinline__ v16bf ld16(const bf16* p0, const bf16* p1) {
  v8bf a = *(const v8bf*)p0;
  v8bf b = *(const v8bf*)p1;
  v16bf r;
#pragma unroll
  for (int i = 0; i < 8; ++i) { r[i] = a[i]; r[i + 8] = b[i]; }
  return r;
}

__device__ __forceinline__ v8f wmma_bf16(v16bf a, v16bf b, v8f c) {
  // D = A(16x32 bf16) * B(32x16 bf16) + C(16x16 f32)
  return __builtin_amdgcn_wmma_f32_16x16x32_bf16(
      /*neg_a=*/false, a, /*neg_b=*/false, b,
      /*c_mod=*/(short)0, c, /*reuse_a=*/false, /*reuse_b=*/false);
}

// ---------------------------------------------------------------------------
// Kernel 1: convert all weight tensors f32 -> bf16 (E*E == H*E*D == 1M elems)
// ---------------------------------------------------------------------------
__global__ void mha_convert_weights(const float* __restrict__ wq,
                                    const float* __restrict__ wk,
                                    const float* __restrict__ wv,
                                    const float* __restrict__ fcw,
                                    bf16* __restrict__ wqb, bf16* __restrict__ wkb,
                                    bf16* __restrict__ wvb, bf16* __restrict__ fcwb) {
  for (int i = blockIdx.x * blockDim.x + threadIdx.x; i < HED;
       i += gridDim.x * blockDim.x) {
    wqb[i]  = f2bf(wq[i]);
    wkb[i]  = f2bf(wk[i]);
    wvb[i]  = f2bf(wv[i]);
    fcwb[i] = f2bf(fcw[i]);
  }
}

// ---------------------------------------------------------------------------
// Kernel 2: per-head QKV projection.
//   out[b,h,s,d] = sum_e act[b,s,e] * W[h,e,d]   (act f32, W bf16, out bf16)
// Block: 256 thr (8 waves). Tile M=128, N=64(=D). Wave tile 32x32 (2x2 WMMA).
// grid = (M/128, H, 3)  with z selecting q/k/v.
// ---------------------------------------------------------------------------
#define LDA 40  // 32 + 8 pad (keeps 16B alignment, avoids bank conflicts)

__global__ __launch_bounds__(256) void mha_qkv_proj(
    const float* __restrict__ qa, const float* __restrict__ ka,
    const float* __restrict__ va,
    const bf16* __restrict__ wqb, const bf16* __restrict__ wkb,
    const bf16* __restrict__ wvb,
    bf16* __restrict__ Qo, bf16* __restrict__ Ko, bf16* __restrict__ Vo) {
  const int zi = blockIdx.z;
  const float* A = (zi == 0) ? qa : (zi == 1) ? ka : va;
  const bf16*  W = (zi == 0) ? wqb : (zi == 1) ? wkb : wvb;
  bf16*        O = (zi == 0) ? Qo : (zi == 1) ? Ko : Vo;

  const int h  = blockIdx.y;
  const int m0 = blockIdx.x * 128;

  __shared__ bf16 At[128 * LDA];
  __shared__ bf16 Bt[64 * LDA];

  const int tid  = threadIdx.x;
  const int lane = tid & 31;
  const int wv32 = tid >> 5;
  const int wm   = (wv32 & 3) * 32;
  const int wn   = (wv32 >> 2) * 32;
  const int half = lane >> 4;
  const int l15  = lane & 15;

  v8f acc[2][2];
#pragma unroll
  for (int i = 0; i < 2; ++i)
#pragma unroll
    for (int j = 0; j < 2; ++j) acc[i][j] = vzero();

  const bf16* Wh = W + (size_t)h * Ee * Dd;
  const int arow = tid >> 1;
  const int ac0  = (tid & 1) * 16;

  for (int kb = 0; kb < Ee; kb += 32) {
    // ---- stage A (f32 -> bf16), 128x32 tile ----
    {
      const float* src = A + (size_t)(m0 + arow) * Ee + kb + ac0;
      if (kb + 32 < Ee) __builtin_prefetch(src + 32, 0, 1);
#pragma unroll
      for (int i = 0; i < 16; ++i) At[arow * LDA + ac0 + i] = f2bf(src[i]);
    }
    // ---- stage B transposed into [n][k] (source is [k][n], row-major ExD) ----
#pragma unroll
    for (int i = 0; i < 8; ++i) {
      int idx = tid * 8 + i;
      int n   = idx & 63;
      int kk  = idx >> 6;
      Bt[n * LDA + kk] = Wh[(size_t)(kb + kk) * Dd + n];
    }
    __syncthreads();

    v16bf a[2], bop[2];
#pragma unroll
    for (int mi = 0; mi < 2; ++mi) {
      const bf16* base = &At[(wm + mi * 16 + l15) * LDA + (half ? 8 : 0)];
      a[mi] = ld16(base, base + 16);
    }
#pragma unroll
    for (int ni = 0; ni < 2; ++ni) {
      const bf16* base = &Bt[(wn + ni * 16 + l15) * LDA + (half ? 8 : 0)];
      bop[ni] = ld16(base, base + 16);
    }
#pragma unroll
    for (int mi = 0; mi < 2; ++mi)
#pragma unroll
      for (int ni = 0; ni < 2; ++ni)
        acc[mi][ni] = wmma_bf16(a[mi], bop[ni], acc[mi][ni]);
    __syncthreads();
  }

  // ---- epilogue: write bf16 to [b,h,s,d] layout ----
#pragma unroll
  for (int mi = 0; mi < 2; ++mi)
#pragma unroll
    for (int ni = 0; ni < 2; ++ni)
#pragma unroll
      for (int r = 0; r < 8; ++r) {
        int row = m0 + wm + mi * 16 + r + 8 * half;  // b*S + s
        int col = wn + ni * 16 + l15;                // d
        int bi  = row >> 11;
        int si  = row & (Ss - 1);
        O[((size_t)(bi * Hh + h) * Ss + si) * Dd + col] = f2bf(acc[mi][ni][r]);
      }
}

// ---------------------------------------------------------------------------
// Kernel 3: flash attention per (b, h, 64-row query tile).
// Block: 128 thr (4 waves); each wave owns 16 query rows.
// K tile staged row-major via async global->LDS DMA (B operand for scores);
// V tile transposed on the VALU path (B operand for PV); P round-trips
// per-wave LDS to reach A-operand layout.
// ---------------------------------------------------------------------------
#define FLD 72  // 64 + 8 pad

__global__ __launch_bounds__(128) void mha_flash_attn(
    const bf16* __restrict__ Q, const bf16* __restrict__ K,
    const bf16* __restrict__ V, bf16* __restrict__ ctx) {
  const int qt  = blockIdx.x;  // query tile (0..31)
  const int h   = blockIdx.y;
  const int b   = blockIdx.z;
  const int tid = threadIdx.x;
  const int lane = tid & 31;
  const int w    = tid >> 5;  // wave 0..3
  const int half = lane >> 4;
  const int l15  = lane & 15;

  __shared__ bf16 Kt[64 * FLD];
  __shared__ bf16 Vt[64 * FLD];
  __shared__ bf16 Pb[4][16 * FLD];

  const size_t bh = (size_t)(b * Hh + h) * Ss * Dd;
  const bf16* Qg = Q + bh + (size_t)qt * 64 * Dd;
  const bf16* Kg = K + bh;
  const bf16* Vg = V + bh;

  const int srow = tid >> 1;
  const int sc0  = (tid & 1) * 32;

  // ---- stage Q tile through Kt (async DMA), pull A operands into regs ----
#pragma unroll
  for (int i = 0; i < 32; i += 8)
    async_cp16(&Qg[srow * 64 + sc0 + i], &Kt[srow * FLD + sc0 + i]);
  async_wait();
  __syncthreads();
  v16bf aq[2];
#pragma unroll
  for (int kc = 0; kc < 2; ++kc) {
    const bf16* base = &Kt[(w * 16 + l15) * FLD + kc * 32 + (half ? 8 : 0)];
    aq[kc] = ld16(base, base + 16);
  }
  __syncthreads();

  v8f acco[4];
#pragma unroll
  for (int j = 0; j < 4; ++j) acco[j] = vzero();
  float mrun[8], lrun[8];
#pragma unroll
  for (int r = 0; r < 8; ++r) { mrun[r] = -1e30f; lrun[r] = 0.0f; }

  const float scale = 0.125f;  // 1/sqrt(64)

  for (int t0 = 0; t0 < Ss; t0 += 64) {
    // ---- stage K (async DMA, row-major) and V (VALU transpose) tiles ----
    {
      const bf16* ks = &Kg[(size_t)(t0 + srow) * 64 + sc0];
#pragma unroll
      for (int i = 0; i < 32; i += 8)
        async_cp16(&ks[i], &Kt[srow * FLD + sc0 + i]);
      const bf16* vs = &Vg[(size_t)(t0 + srow) * 64 + sc0];
      if (t0 + 64 < Ss) __builtin_prefetch(vs + (size_t)64 * 64, 0, 1);
#pragma unroll
      for (int i = 0; i < 32; i += 8) {
        uint4 dv = *(const uint4*)&vs[i];
        const bf16* pv = (const bf16*)&dv;
#pragma unroll
        for (int j2 = 0; j2 < 8; ++j2) Vt[(sc0 + i + j2) * FLD + srow] = pv[j2];
      }
    }
    async_wait();
    __syncthreads();

    // ---- scores: 16x64 = Q(16x64) x K^T, 8 WMMA ----
    v8f sc[4];
#pragma unroll
    for (int j = 0; j < 4; ++j) sc[j] = vzero();
#pragma unroll
    for (int kc = 0; kc < 2; ++kc)
#pragma unroll
      for (int j = 0; j < 4; ++j) {
        const bf16* base = &Kt[(j * 16 + l15) * FLD + kc * 32 + (half ? 8 : 0)];
        v16bf bk = ld16(base, base + 16);
        sc[j] = wmma_bf16(aq[kc], bk, sc[j]);
      }

    // ---- online softmax (rows live at (vgpr r, lane half)) ----
    float corr[8], lt[8];
#pragma unroll
    for (int r = 0; r < 8; ++r) {
      float m = fmaxf(fmaxf(sc[0][r], sc[1][r]), fmaxf(sc[2][r], sc[3][r]));
#pragma unroll
      for (int off = 1; off <= 8; off <<= 1) m = fmaxf(m, __shfl_xor(m, off, 32));
      float mnew = fmaxf(mrun[r], m * scale);
      corr[r] = __expf(mrun[r] - mnew);
      mrun[r] = mnew;
      lt[r] = 0.0f;
    }
#pragma unroll
    for (int j = 0; j < 4; ++j)
#pragma unroll
      for (int r = 0; r < 8; ++r) acco[j][r] *= corr[r];
#pragma unroll
    for (int j = 0; j < 4; ++j)
#pragma unroll
      for (int r = 0; r < 8; ++r) {
        float p = __expf(sc[j][r] * scale - mrun[r]);
        lt[r] += p;
        Pb[w][(r + 8 * half) * FLD + j * 16 + l15] = f2bf(p);
      }
#pragma unroll
    for (int r = 0; r < 8; ++r) {
      float s = lt[r];
#pragma unroll
      for (int off = 1; off <= 8; off <<= 1) s += __shfl_xor(s, off, 32);
      lrun[r] = lrun[r] * corr[r] + s;
    }

    // ---- PV: ctx(16x64) += P(16x64) x V(64x64), 8 WMMA ----
#pragma unroll
    for (int kc = 0; kc < 2; ++kc) {
      const bf16* pbase = &Pb[w][l15 * FLD + kc * 32 + (half ? 8 : 0)];
      v16bf ap = ld16(pbase, pbase + 16);
#pragma unroll
      for (int j = 0; j < 4; ++j) {
        const bf16* vbase = &Vt[(j * 16 + l15) * FLD + kc * 32 + (half ? 8 : 0)];
        v16bf bv = ld16(vbase, vbase + 16);
        acco[j] = wmma_bf16(ap, bv, acco[j]);
      }
    }
    __syncthreads();
  }

  // ---- epilogue: normalize, write ctx bf16 in [B,S,H*D] ----
#pragma unroll
  for (int r = 0; r < 8; ++r) lrun[r] = 1.0f / lrun[r];
#pragma unroll
  for (int j = 0; j < 4; ++j)
#pragma unroll
    for (int r = 0; r < 8; ++r) {
      int si = qt * 64 + w * 16 + r + 8 * half;
      int di = j * 16 + l15;
      ctx[((size_t)b * Ss + si) * Ee + h * Dd + di] = f2bf(acco[j][r] * lrun[r]);
    }
}

// ---------------------------------------------------------------------------
// Kernel 4: output projection  out = ctx @ fc_w.T + fc_b   (f32 out)
// B[k=e][n] = fc_w[n][e]  -> bf16 source is already [n][k]; both tiles are
// pure copies, staged via async global->LDS DMA.
// grid = (M/128, E/64), block 256.
// ---------------------------------------------------------------------------
__global__ __launch_bounds__(256) void mha_out_proj(
    const bf16* __restrict__ ctx, const bf16* __restrict__ fcwb,
    const float* __restrict__ fcb, float* __restrict__ out) {
  const int m0 = blockIdx.x * 128;
  const int n0 = blockIdx.y * 64;

  __shared__ bf16 At[128 * LDA];
  __shared__ bf16 Bt[64 * LDA];

  const int tid  = threadIdx.x;
  const int lane = tid & 31;
  const int wv32 = tid >> 5;
  const int wm   = (wv32 & 3) * 32;
  const int wn   = (wv32 >> 2) * 32;
  const int half = lane >> 4;
  const int l15  = lane & 15;

  v8f acc[2][2];
#pragma unroll
  for (int i = 0; i < 2; ++i)
#pragma unroll
    for (int j = 0; j < 2; ++j) acc[i][j] = vzero();

  const int arow = tid >> 1;
  const int ac0  = (tid & 1) * 16;
  const int brow = tid >> 2;
  const int bc0  = (tid & 3) * 8;

  for (int kb = 0; kb < Ee; kb += 32) {
    {
      const bf16* src = ctx + (size_t)(m0 + arow) * Ee + kb + ac0;
      if (kb + 32 < Ee) __builtin_prefetch(src + 32, 0, 1);
      async_cp16(src,     &At[arow * LDA + ac0]);
      async_cp16(src + 8, &At[arow * LDA + ac0 + 8]);
    }
    async_cp16(&fcwb[(size_t)(n0 + brow) * Ee + kb + bc0], &Bt[brow * LDA + bc0]);
    async_wait();
    __syncthreads();

    v16bf a[2], bop[2];
#pragma unroll
    for (int mi = 0; mi < 2; ++mi) {
      const bf16* base = &At[(wm + mi * 16 + l15) * LDA + (half ? 8 : 0)];
      a[mi] = ld16(base, base + 16);
    }
#pragma unroll
    for (int ni = 0; ni < 2; ++ni) {
      const bf16* base = &Bt[(wn + ni * 16 + l15) * LDA + (half ? 8 : 0)];
      bop[ni] = ld16(base, base + 16);
    }
#pragma unroll
    for (int mi = 0; mi < 2; ++mi)
#pragma unroll
      for (int ni = 0; ni < 2; ++ni)
        acc[mi][ni] = wmma_bf16(a[mi], bop[ni], acc[mi][ni]);
    __syncthreads();
  }

#pragma unroll
  for (int mi = 0; mi < 2; ++mi)
#pragma unroll
    for (int ni = 0; ni < 2; ++ni)
#pragma unroll
      for (int r = 0; r < 8; ++r) {
        int row = m0 + wm + mi * 16 + r + 8 * half;
        int col = n0 + wn + ni * 16 + l15;
        out[(size_t)row * Ee + col] = acc[mi][ni][r] + fcb[col];
      }
}

// ---------------------------------------------------------------------------
extern "C" void kernel_launch(void* const* d_in, const int* in_sizes, int n_in,
                              void* d_out, int out_size, void* d_ws, size_t ws_size,
                              hipStream_t stream) {
  const float* q   = (const float*)d_in[0];
  const float* k   = (const float*)d_in[1];
  const float* v   = (const float*)d_in[2];
  const float* wq  = (const float*)d_in[3];
  const float* wk  = (const float*)d_in[4];
  const float* wvp = (const float*)d_in[5];
  const float* fcw = (const float*)d_in[6];
  const float* fcb = (const float*)d_in[7];
  float* out = (float*)d_out;

  char* ws = (char*)d_ws;
  size_t off = 0;
  auto carve = [&](size_t bytes) -> void* {
    void* p = ws + off;
    off += (bytes + 255) & ~(size_t)255;
    return p;
  };
  bf16* wqb  = (bf16*)carve((size_t)HED * 2);
  bf16* wkb  = (bf16*)carve((size_t)HED * 2);
  bf16* wvb  = (bf16*)carve((size_t)HED * 2);
  bf16* fcwb = (bf16*)carve((size_t)Ee * Ee * 2);
  bf16* Qb   = (bf16*)carve(BHSD * 2);
  bf16* Kb   = (bf16*)carve(BHSD * 2);
  bf16* Vb   = (bf16*)carve(BHSD * 2);
  bf16* ctxb = (bf16*)carve(BSE * 2);

  mha_convert_weights<<<512, 256, 0, stream>>>(wq, wk, wvp, fcw, wqb, wkb, wvb, fcwb);
  mha_qkv_proj<<<dim3(Mm / 128, Hh, 3), 256, 0, stream>>>(q, k, v, wqb, wkb, wvb,
                                                          Qb, Kb, Vb);
  mha_flash_attn<<<dim3(Ss / 64, Hh, Bb), 128, 0, stream>>>(Qb, Kb, Vb, ctxb);
  mha_out_proj<<<dim3(Mm / 128, Ee / 64), 256, 0, stream>>>(ctxb, fcwb, fcb, out);
}
